// SS2D_77163382440854
// MI455X (gfx1250) — matrix operand order repaired and checked by reading, not compiled, not run on hardware
//
#include <hip/hip_runtime.h>

// ---------------------------------------------------------------------------
// SS2D (VMamba selective-scan 2D block) for gfx1250 (MI455X).
//  - Dense projections via v_wmma_f32_16x16x32_f16 (wave32 WMMA).
//  - Selective scan: segmented two-pass scan (exp-of-sum transition collapse)
//    to convert the latency-bound sequential recurrence into a
//    throughput-bound computation across the whole chip.
//  - Scan operand staging uses gfx1250 async global->LDS copies
//    (global_load_async_to_lds_b128 + s_wait_asynccnt), no VGPR bounce.
// ---------------------------------------------------------------------------

typedef __attribute__((ext_vector_type(16))) _Float16 v16h;
typedef __attribute__((ext_vector_type(8)))  float    v8f;

#define B_     8
#define H_     64
#define W_     64
#define C_     96
#define Di_    192
#define N_     16
#define R_     6
#define K_     4
#define L_     (H_ * W_)        // 4096
#define M_     (B_ * L_)        // 32768
#define CPROJ  (R_ + 2 * N_)    // 38
#define KC_    (K_ * Di_)       // 768
#define NCH    (B_ * KC_)       // 6144 scan channels
#define S_     32               // scan segments
#define SEGLEN (L_ / S_)        // 128
#define CHUNK  64               // LDS staging chunk (2 per segment)

// --- WMMA per-lane K index maps (CDNA5 ISA 7.12.2, 16-bit operands) --------
// A 16x32: lanes 0-15 hold K {0..7,16..23}; lanes 16-31 hold K {8..15,24..31}
__device__ __forceinline__ int ka_map(int e, int lane) {
  return e + (e & 8) + ((lane & 16) >> 1);
}
// B 32x16: lanes 0-15 hold K {0..7,16..23}; lanes 16-31 hold K {8..15,24..31}
__device__ __forceinline__ int kb_map(int e, int lane) {
  return e + (e & 8) + ((lane & 16) >> 1);
}

// Cross-scan index: xs[b,k,d,l] -> offset into xc stored as (B,Di,H,W)
__device__ __forceinline__ size_t xs_idx(int b, int k, int d, int l) {
  int ll = (k & 2) ? (L_ - 1 - l) : l;     // k=2,3: reversed
  int pos;
  if (k & 1) {                             // k=1,3: H/W transposed scan order
    int ww = ll >> 6;                      // ll / H_   (H_=64)
    int hh = ll & 63;                      // ll % H_
    pos = hh * W_ + ww;
  } else {
    pos = ll;
  }
  return (size_t)(b * Di_ + d) * L_ + pos;
}

__device__ __forceinline__ float silu_f(float v) {
  return v / (1.f + __expf(-v));
}

// ---------------------------------------------------------------------------
// Kernel 1: xz[m, 0:384] = x[m, 0:96] @ in_proj_w^T        (WMMA f16)
// grid (2048, 6), block 128 (4 waves; wave -> one 16x16 N tile)
// ---------------------------------------------------------------------------
__global__ void __launch_bounds__(128)
gemm_inproj(const float* __restrict__ x, const float* __restrict__ w,
            float* __restrict__ xz) {
  const int lane = threadIdx.x & 31;
  const int wave = threadIdx.x >> 5;
  const int mt = blockIdx.x;                 // 0..2047
  const int nt = blockIdx.y * 4 + wave;      // 0..23
  const int row = mt * 16 + (lane & 15);
  const int col = nt * 16 + (lane & 15);
  v8f acc = {};
  for (int kk = 0; kk < C_; kk += 32) {
    v16h a, bf;
    const float* xa = x + (size_t)row * C_ + kk;
    const float* wb = w + (size_t)col * C_ + kk;
#pragma unroll
    for (int e = 0; e < 16; ++e) {
      a[e]  = (_Float16)xa[ka_map(e, lane)];
      bf[e] = (_Float16)wb[kb_map(e, lane)];
    }
    acc = __builtin_amdgcn_wmma_f32_16x16x32_f16(false, a, false, bf,
                                                 (short)0, acc, false, false);
  }
  const int mbase = mt * 16 + ((lane & 16) >> 1);
#pragma unroll
  for (int r = 0; r < 8; ++r)
    xz[(size_t)(mbase + r) * (2 * Di_) + col] = acc[r];
}

// ---------------------------------------------------------------------------
// Kernel 2: depthwise 3x3 SAME conv over xp (first Di channels of xz) + SiLU
// output xc in (B, Di, H, W) layout (L-contiguous for the scan)
// ---------------------------------------------------------------------------
__global__ void __launch_bounds__(256)
conv_silu(const float* __restrict__ xz, const float* __restrict__ cw,
          const float* __restrict__ cb, float* __restrict__ xc) {
  int idx = blockIdx.x * blockDim.x + threadIdx.x;
  if (idx >= B_ * Di_ * L_) return;
  int wq = idx & 63;
  int t  = idx >> 6;
  int hq = t & 63;
  t >>= 6;
  int d = t % Di_;
  int b = t / Di_;
  float s = 0.f;
#pragma unroll
  for (int kh = 0; kh < 3; ++kh) {
    int hh = hq + kh - 1;
    if ((unsigned)hh >= H_) continue;
#pragma unroll
    for (int kw = 0; kw < 3; ++kw) {
      int ww = wq + kw - 1;
      if ((unsigned)ww >= W_) continue;
      s += xz[((size_t)((b * H_ + hh) * W_) + ww) * (2 * Di_) + d] *
           cw[d * 9 + kh * 3 + kw];
    }
  }
  s += cb[d];
  xc[idx] = silu_f(s);
}

// ---------------------------------------------------------------------------
// Kernel 3: x_dbl[b,k,c,l] = sum_d x_proj_w[k,c,d] * xs[b,k,d,l]   (WMMA f16)
// cross-scan folded into the B-operand gathers (xc is 25MB -> lives in L2)
// grid (64, 3, 32), block 128
// ---------------------------------------------------------------------------
__global__ void __launch_bounds__(128)
gemm_xdbl(const float* __restrict__ xc, const float* __restrict__ xpw,
          float* __restrict__ xdbl) {
  const int lane = threadIdx.x & 31;
  const int wave = threadIdx.x >> 5;
  const int nt = blockIdx.x * 4 + wave;   // 0..255 (L tiles)
  const int mt = blockIdx.y;              // 0..2   (38 rows -> 3 tiles)
  const int bk = blockIdx.z;              // 0..31
  const int b = bk >> 2, k = bk & 3;
  const int crow = mt * 16 + (lane & 15);
  const int l    = nt * 16 + (lane & 15);
  const float* wk = xpw + (size_t)k * CPROJ * Di_;
  v8f acc = {};
  for (int kk = 0; kk < Di_; kk += 32) {
    v16h a, bf;
#pragma unroll
    for (int e = 0; e < 16; ++e) {
      int kaa = kk + ka_map(e, lane);
      a[e] = (crow < CPROJ) ? (_Float16)wk[(size_t)crow * Di_ + kaa]
                            : (_Float16)0.f;
      int kbb = kk + kb_map(e, lane);
      bf[e] = (_Float16)xc[xs_idx(b, k, kbb, l)];
    }
    acc = __builtin_amdgcn_wmma_f32_16x16x32_f16(false, a, false, bf,
                                                 (short)0, acc, false, false);
  }
  const int cbase = mt * 16 + ((lane & 16) >> 1);
#pragma unroll
  for (int r = 0; r < 8; ++r) {
    int c = cbase + r;
    if (c < CPROJ)
      xdbl[((size_t)bk * CPROJ + c) * L_ + l] = acc[r];
  }
}

// ---------------------------------------------------------------------------
// Kernels 4a/4c: segmented selective scan (templated on pass).
//
// The diagonal linear recurrence h_l = exp(d_l*A)*h_{l-1} + d_l*B_l*u_l has a
// segment transition that collapses to exp(A * sum(d_l)) (product of exps).
// Pass 1 (!FINAL): each of S_=32 segments scans locally from h=0, emitting
//   sum(delta) and h_end per channel.  grid (S_, 32) -> 6144 waves device-wide
//   (vs 192 for a monolithic scan) => throughput-bound, not latency-bound.
// Pass 3 (FINAL): identical local scan seeded with the carry-in from the
//   fixup pass; writes y (+ u*D skip term).
// One block per (segment, b, k); 192 threads = one scan channel d each.
// h[16]/A[16] in VGPRs; shared dt-raw/Bs/Cs staged in LDS per 64-step chunk
// via gfx1250 async global->LDS copies (ASYNCcnt-fenced, no VGPR bounce).
// ---------------------------------------------------------------------------
template <bool FINAL>
__global__ void __launch_bounds__(192)
scan_pass(const float* __restrict__ xc, const float* __restrict__ xdbl,
          const float* __restrict__ dtw, const float* __restrict__ dtb,
          const float* __restrict__ Alogs, const float* __restrict__ Dskip,
          const float* __restrict__ hcarry,  // [S][NCH][16]  (FINAL)
          float* __restrict__ sdelta,        // [S][NCH]      (!FINAL)
          float* __restrict__ hend,          // [S][NCH][16]  (!FINAL)
          float* __restrict__ y) {           //               (FINAL)
  __shared__ float sh[CPROJ][CHUNK];   // rows 0..5 dt-raw, 6..21 Bs, 22..37 Cs
  const int seg = blockIdx.x;          // 0..S_-1
  const int bk  = blockIdx.y;          // 0..31
  const int b = bk >> 2, k = bk & 3;
  const int d = threadIdx.x;           // 0..191
  const int kc = k * Di_ + d;
  const int ch = bk * Di_ + d;         // global channel 0..6143

  float wdt[R_];
#pragma unroll
  for (int r = 0; r < R_; ++r) wdt[r] = dtw[(size_t)kc * R_ + r];
  const float bias  = dtb[kc];
  const float Dterm = Dskip[kc];
  float Av[N_];
#pragma unroll
  for (int n = 0; n < N_; ++n) Av[n] = -__expf(Alogs[(size_t)kc * N_ + n]);

  float h[N_];
  if (FINAL) {
    const float* hc = hcarry + ((size_t)seg * NCH + ch) * N_;
#pragma unroll
    for (int n = 0; n < N_; ++n) h[n] = hc[n];
  } else {
#pragma unroll
    for (int n = 0; n < N_; ++n) h[n] = 0.f;
  }
  float sd = 0.f;

  const float* src  = xdbl + (size_t)bk * CPROJ * L_;
  float* yout = y + (size_t)(bk * Di_ + d) * L_;
  const int lbase = seg * SEGLEN;

  for (int l0 = lbase; l0 < lbase + SEGLEN; l0 += CHUNK) {
    __syncthreads();
    // Async global->LDS staging: 16B granules, per-lane LDS offset in VDST.
    // (low 32 bits of a generic pointer to a __shared__ object are the
    //  wave-relative LDS byte offset; rows are 256B so 16B alignment holds)
    for (int t = threadIdx.x; t < (CPROJ * CHUNK) / 4; t += 192) {
      int c = t / (CHUNK / 4), q = t % (CHUNK / 4);
      unsigned lds_off = (unsigned)(size_t)&sh[c][q * 4];
      const float* gp = src + (size_t)c * L_ + l0 + q * 4;
      asm volatile("global_load_async_to_lds_b128 %0, %1, off"
                   :: "v"(lds_off), "v"(gp) : "memory");
    }
    asm volatile("s_wait_asynccnt 0" ::: "memory");
    __syncthreads();
    if (l0 + CHUNK < lbase + SEGLEN && threadIdx.x < CPROJ)
      __builtin_prefetch(src + (size_t)threadIdx.x * L_ + l0 + CHUNK, 0, 1);

    for (int j = 0; j < CHUNK; ++j) {
      const int l = l0 + j;
      float raw = bias;
#pragma unroll
      for (int r = 0; r < R_; ++r) raw += wdt[r] * sh[r][j];
      float delta = (raw > 20.f) ? raw : __logf(1.f + __expf(raw));
      float u  = xc[xs_idx(b, k, d, l)];
      float du = delta * u;
      float acc = 0.f;
#pragma unroll
      for (int n = 0; n < N_; ++n) {
        float dA = __expf(delta * Av[n]);
        h[n] = dA * h[n] + du * sh[R_ + n][j];
        acc += h[n] * sh[R_ + N_ + n][j];
      }
      if (FINAL) yout[l] = acc + u * Dterm;
      else       sd += delta;
    }
  }

  if (!FINAL) {
    sdelta[(size_t)seg * NCH + ch] = sd;
    float* he = hend + ((size_t)seg * NCH + ch) * N_;
#pragma unroll
    for (int n = 0; n < N_; ++n) he[n] = h[n];
  }
}

// ---------------------------------------------------------------------------
// Kernel 4b: inter-segment fixup.  Per channel, sequentially combine the 32
// segment summaries: carry(s) = h_in, h_in' = exp(A*sum_delta)*h_in + h_end.
// 6144 threads total; 32 trivial steps.
// ---------------------------------------------------------------------------
__global__ void __launch_bounds__(256)
scan_fixup(const float* __restrict__ Alogs, const float* __restrict__ sdelta,
           const float* __restrict__ hend, float* __restrict__ hcarry) {
  int ch = blockIdx.x * blockDim.x + threadIdx.x;
  if (ch >= NCH) return;
  const int k  = (ch / Di_) & 3;
  const int d  = ch % Di_;
  const int kc = k * Di_ + d;
  float Av[N_];
#pragma unroll
  for (int n = 0; n < N_; ++n) Av[n] = -__expf(Alogs[(size_t)kc * N_ + n]);
  float h[N_];
#pragma unroll
  for (int n = 0; n < N_; ++n) h[n] = 0.f;
  for (int s = 0; s < S_; ++s) {
    float* hc = hcarry + ((size_t)s * NCH + ch) * N_;
    const float* he = hend + ((size_t)s * NCH + ch) * N_;
    const float sd = sdelta[(size_t)s * NCH + ch];
#pragma unroll
    for (int n = 0; n < N_; ++n) {
      hc[n] = h[n];
      h[n] = __expf(Av[n] * sd) * h[n] + he[n];
    }
  }
}

// ---------------------------------------------------------------------------
// Kernel 5: cross-merge of the 4 direction outputs -> merged (B, Di, H, W)
// ---------------------------------------------------------------------------
__global__ void __launch_bounds__(256)
merge_kernel(const float* __restrict__ y, float* __restrict__ merged) {
  int idx = blockIdx.x * blockDim.x + threadIdx.x;
  if (idx >= B_ * Di_ * L_) return;
  int pos = idx % L_;
  int bd  = idx / L_;
  int b = bd / Di_, d = bd % Di_;
  int hq = pos >> 6, wq = pos & 63;
  int lt = wq * H_ + hq;                  // transposed scan position
  const float* yb = y + ((size_t)(b * K_) * Di_ + d) * L_;
  const size_t ks = (size_t)Di_ * L_;     // stride between k
  float v = yb[pos]
          + yb[2 * ks + (L_ - 1 - pos)]
          + yb[1 * ks + lt]
          + yb[3 * ks + (L_ - 1 - lt)];
  merged[idx] = v;
}

// ---------------------------------------------------------------------------
// Kernel 6: LayerNorm over Di + SiLU(z) gate; emits f16 activations for the
// final WMMA GEMM. One wave32 per pixel, 8 pixels per 256-thread block.
// ---------------------------------------------------------------------------
__global__ void __launch_bounds__(256)
ln_gate(const float* __restrict__ merged, const float* __restrict__ xz,
        const float* __restrict__ g, const float* __restrict__ bb,
        _Float16* __restrict__ t) {
  const int lane = threadIdx.x & 31;
  const int wave = threadIdx.x >> 5;
  const int m = blockIdx.x * 8 + wave;    // pixel index 0..32767
  const int b = m / L_, pos = m % L_;
  const float* mb = merged + (size_t)b * Di_ * L_ + pos;  // element d at mb[d*L_]
  float v[6];
  float s = 0.f, s2 = 0.f;
#pragma unroll
  for (int i = 0; i < 6; ++i) {
    int d = lane + 32 * i;
    v[i] = mb[(size_t)d * L_];
    s  += v[i];
    s2 += v[i] * v[i];
  }
#pragma unroll
  for (int off = 16; off > 0; off >>= 1) {
    s  += __shfl_down(s, off, 32);
    s2 += __shfl_down(s2, off, 32);
  }
  s  = __shfl(s, 0, 32);
  s2 = __shfl(s2, 0, 32);
  const float mean = s / Di_;
  const float var  = s2 / Di_ - mean * mean;
  const float inv  = rsqrtf(var + 1e-5f);
  const float* zrow = xz + (size_t)m * (2 * Di_) + Di_;
#pragma unroll
  for (int i = 0; i < 6; ++i) {
    int d = lane + 32 * i;
    float val = (v[i] - mean) * inv * g[d] + bb[d];
    t[(size_t)m * Di_ + d] = (_Float16)(val * silu_f(zrow[d]));
  }
}

// ---------------------------------------------------------------------------
// Kernel 7: out[m, 0:96] = t[m, 0:192] @ out_proj_w^T       (WMMA f16)
// grid 2048, block 192 (6 waves; wave -> one of 6 N tiles)
// ---------------------------------------------------------------------------
__global__ void __launch_bounds__(192)
gemm_out(const _Float16* __restrict__ t, const float* __restrict__ wout,
         float* __restrict__ out) {
  const int lane = threadIdx.x & 31;
  const int nt = threadIdx.x >> 5;        // 0..5
  const int mt = blockIdx.x;              // 0..2047
  const int row = mt * 16 + (lane & 15);
  const int col = nt * 16 + (lane & 15);
  v8f acc = {};
  for (int kk = 0; kk < Di_; kk += 32) {
    v16h a, bf;
    const _Float16* ta = t + (size_t)row * Di_ + kk;
    const float* wb = wout + (size_t)col * Di_ + kk;
#pragma unroll
    for (int e = 0; e < 16; ++e) {
      a[e]  = ta[ka_map(e, lane)];
      bf[e] = (_Float16)wb[kb_map(e, lane)];
    }
    acc = __builtin_amdgcn_wmma_f32_16x16x32_f16(false, a, false, bf,
                                                 (short)0, acc, false, false);
  }
  const int mbase = mt * 16 + ((lane & 16) >> 1);
#pragma unroll
  for (int r = 0; r < 8; ++r)
    out[(size_t)(mbase + r) * C_ + col] = acc[r];
}

// ---------------------------------------------------------------------------
extern "C" void kernel_launch(void* const* d_in, const int* in_sizes, int n_in,
                              void* d_out, int out_size, void* d_ws, size_t ws_size,
                              hipStream_t stream) {
  const float* x          = (const float*)d_in[0];   // (B,H,W,C)
  const float* in_proj_w  = (const float*)d_in[1];   // (2Di, C)
  const float* conv_w     = (const float*)d_in[2];   // (Di,1,3,3)
  const float* conv_b     = (const float*)d_in[3];   // (Di,)
  const float* x_proj_w   = (const float*)d_in[4];   // (K, 38, Di)
  const float* dt_projs_w = (const float*)d_in[5];   // (K, Di, R)
  const float* dt_projs_b = (const float*)d_in[6];   // (K, Di)
  const float* A_logs     = (const float*)d_in[7];   // (KC, N)
  const float* Ds         = (const float*)d_in[8];   // (KC,)
  const float* norm_g     = (const float*)d_in[9];   // (Di,)
  const float* norm_b     = (const float*)d_in[10];  // (Di,)
  const float* out_proj_w = (const float*)d_in[11];  // (C, Di)

  float* ws     = (float*)d_ws;
  float* xz     = ws;                                    // M * 2Di
  float* xc     = xz + (size_t)M_ * 2 * Di_;             // B*Di*L
  float* xdbl   = xc + (size_t)B_ * Di_ * L_;            // B*K*38*L
  float* yscan  = xdbl + (size_t)B_ * K_ * CPROJ * L_;   // B*K*Di*L
  float* sdelta = yscan + (size_t)B_ * K_ * Di_ * L_;    // S*NCH
  float* hend   = sdelta + (size_t)S_ * NCH;             // S*NCH*16
  float* hcarry = hend + (size_t)S_ * NCH * N_;          // S*NCH*16
  float* merged = xc;                                    // reuse xc after scan
  _Float16* tbuf = (_Float16*)xdbl;                      // reuse xdbl after scan

  gemm_inproj<<<dim3(2048, 6), 128, 0, stream>>>(x, in_proj_w, xz);
  conv_silu<<<(B_ * Di_ * L_ + 255) / 256, 256, 0, stream>>>(xz, conv_w, conv_b, xc);
  gemm_xdbl<<<dim3(64, 3, 32), 128, 0, stream>>>(xc, x_proj_w, xdbl);

  scan_pass<false><<<dim3(S_, B_ * K_), 192, 0, stream>>>(
      xc, xdbl, dt_projs_w, dt_projs_b, A_logs, Ds,
      nullptr, sdelta, hend, nullptr);
  scan_fixup<<<(NCH + 255) / 256, 256, 0, stream>>>(A_logs, sdelta, hend, hcarry);
  scan_pass<true><<<dim3(S_, B_ * K_), 192, 0, stream>>>(
      xc, xdbl, dt_projs_w, dt_projs_b, A_logs, Ds,
      hcarry, nullptr, nullptr, yscan);

  merge_kernel<<<(B_ * Di_ * L_ + 255) / 256, 256, 0, stream>>>(yscan, merged);
  ln_gate<<<M_ / 8, 256, 0, stream>>>(merged, xz, norm_g, norm_b, tbuf);
  gemm_out<<<2048, 192, 0, stream>>>(tbuf, out_proj_w, (float*)d_out);
}
